// BaseChunkLoss_24326694765139
// MI455X (gfx1250) — matrix operand order repaired and checked by reading, not compiled, not run on hardware
//
#include <hip/hip_runtime.h>
#include <hip/hip_bf16.h>

typedef __attribute__((ext_vector_type(16))) __bf16 v16bf;
typedef __attribute__((ext_vector_type(8)))  float  v8f;

#define TOKENS   4096
#define DIM      1024
#define VOCAB    32000
#define NSLICE   4
#define VPS      (VOCAB / NSLICE)      // 8000 vocab rows per slice
#define M_TILE   32
#define NTILES_TOK (TOKENS / M_TILE)   // 128
#define KCHUNKS  (DIM / 32)            // 32 k-chunks of 32
#define NVT      (VOCAB / 16)          // 2000 vocab 16-row tiles
// packed W: per (vocab-16-tile, kchunk): 1024B hi blob + 1024B lo blob
#define WP_BYTES ((size_t)NVT * KCHUNKS * 2048)   // 131,072,000 B

union Frag { uint4 q[2]; v16bf v; };

__device__ __forceinline__ unsigned bf_rne(float x) {
  unsigned u = __builtin_bit_cast(unsigned, x);
  return (u + 0x7fffu + ((u >> 16) & 1u)) >> 16;
}

__device__ __forceinline__ void merge_ms(float& m, float& s, float m2, float s2) {
  float mn = fmaxf(m, m2);
  s = s * __expf(m - mn) + s2 * __expf(m2 - mn);
  m = mn;
}

// ---------------------------------------------------------------------------
// Kernel 1: pack W (f32, row-major [VOCAB][DIM]) into bf16 hi/lo blobs laid
// out exactly in WMMA-B operand order so main-kernel loads are coalesced b128.
// B k order used here, same as A (16-bit, 32x16): lane l -> column n = l%16; K = 16*(l>=16) + j,
// j = 0..15 sequential halves.
// ---------------------------------------------------------------------------
__global__ __launch_bounds__(256) void wprep_kernel(
    const float* __restrict__ W, unsigned char* __restrict__ Wp) {
  int t     = blockIdx.x * 256 + threadIdx.x;
  int seg   = t & 1;
  int lane  = (t >> 1) & 31;
  int kc    = (t >> 6) & 31;
  int ntile = t >> 11;
  if (ntile >= NVT) return;
  int row = ntile * 16 + (lane & 15);
  int kb  = kc * 32 + ((lane >= 16) ? 16 : 0) + seg * 8;
  const float* src = W + (size_t)row * DIM + kb;
  float4 a = *(const float4*)(src);
  float4 b = *(const float4*)(src + 4);
  float v[8] = {a.x, a.y, a.z, a.w, b.x, b.y, b.z, b.w};
  unsigned hi[8], lo[8];
#pragma unroll
  for (int i = 0; i < 8; ++i) {
    unsigned h = bf_rne(v[i]);
    hi[i] = h;
    float hf = __builtin_bit_cast(float, h << 16);
    lo[i] = bf_rne(v[i] - hf);
  }
  uint4 hq = {hi[0] | (hi[1] << 16), hi[2] | (hi[3] << 16),
              hi[4] | (hi[5] << 16), hi[6] | (hi[7] << 16)};
  uint4 lq = {lo[0] | (lo[1] << 16), lo[2] | (lo[3] << 16),
              lo[4] | (lo[5] << 16), lo[6] | (lo[7] << 16)};
  unsigned char* base = Wp + (size_t)(ntile * KCHUNKS + kc) * 2048;
  *(uint4*)(base +        lane * 32 + seg * 16) = hq;
  *(uint4*)(base + 1024 + lane * 32 + seg * 16) = lq;
}

// ---------------------------------------------------------------------------
// Kernel 2: main fused GEMM + online logsumexp.
// grid = (128 token tiles, 4 vocab slices); 256 thr = 8 wave32 (2 Msub x 4 Nsub)
// Two independent accumulator chains (even/odd k-chunks) for WMMA ILP.
// ---------------------------------------------------------------------------
__global__ __launch_bounds__(256) void celoss_main_kernel(
    const float* __restrict__ H, const float* __restrict__ bias,
    const int* __restrict__ labels, const unsigned char* __restrict__ Wp,
    float* __restrict__ m_part, float* __restrict__ s_part,
    float* __restrict__ t_part) {
  __shared__ __align__(16) unsigned char Ahi[2 * 32 * 1024];  // 64 KB
  __shared__ __align__(16) unsigned char Alo[2 * 32 * 1024];  // 64 KB
  __shared__ float stat_m[8 * 32 * 8];                        // 8 KB
  __shared__ float stat_s[8 * 32 * 8];                        // 8 KB
  __shared__ float red_m[256], red_s[256];
  __shared__ float t_lds[32];

  const int tid   = threadIdx.x;
  const int m0    = blockIdx.x * M_TILE;
  const int slice = blockIdx.y;

  if (tid < 32) t_lds[tid] = 0.0f;

  // ---- stage H tile as bf16 hi/lo in WMMA-A operand order -----------------
  // A k order used here, same as B (16-bit, 16x32): lane l -> row l%16; K = (l<16?0:16)+{0..15}
  for (int it = tid; it < 4096; it += 256) {
    int seg = it & 1, lane = (it >> 1) & 31, kc = (it >> 6) & 31, msub = it >> 11;
    int row = m0 + msub * 16 + (lane & 15);
    int kb  = kc * 32 + ((lane >= 16) ? 16 : 0) + seg * 8;
    const float* src = H + (size_t)row * DIM + kb;
    float4 a = *(const float4*)(src);
    float4 b = *(const float4*)(src + 4);
    float v[8] = {a.x, a.y, a.z, a.w, b.x, b.y, b.z, b.w};
    unsigned hi[8], lo[8];
#pragma unroll
    for (int i = 0; i < 8; ++i) {
      unsigned h = bf_rne(v[i]);
      hi[i] = h;
      float hf = __builtin_bit_cast(float, h << 16);
      lo[i] = bf_rne(v[i] - hf);
    }
    uint4 hq = {hi[0] | (hi[1] << 16), hi[2] | (hi[3] << 16),
                hi[4] | (hi[5] << 16), hi[6] | (hi[7] << 16)};
    uint4 lq = {lo[0] | (lo[1] << 16), lo[2] | (lo[3] << 16),
                lo[4] | (lo[5] << 16), lo[6] | (lo[7] << 16)};
    int off = msub * 32768 + kc * 1024 + lane * 32 + seg * 16;
    *(uint4*)(Ahi + off) = hq;
    *(uint4*)(Alo + off) = lq;
  }
  __syncthreads();

  const int wave = tid >> 5, lane = tid & 31;
  const int msub = wave >> 2, nsub = wave & 3;
  const int mrow = (lane < 16) ? 0 : 8;   // C/D: lanes 16-31 hold rows 8..15
  const int col  = lane & 15;

  int lab[8];
#pragma unroll
  for (int i = 0; i < 8; ++i) lab[i] = labels[m0 + msub * 16 + mrow + i];

  float m_run[8], s_run[8];
#pragma unroll
  for (int i = 0; i < 8; ++i) { m_run[i] = -3.0e38f; s_run[i] = 0.0f; }

  const unsigned char* Ah = Ahi + msub * 32768 + lane * 32;
  const unsigned char* Al = Alo + msub * 32768 + lane * 32;

  for (int it2 = 0; it2 < VPS / 64; ++it2) {
    const int nbase = slice * VPS + it2 * 64 + nsub * 16;
    const unsigned char* wb =
        Wp + (size_t)((nbase >> 4) * KCHUNKS) * 2048 + lane * 32;
    v8f acc0 = {0.f, 0.f, 0.f, 0.f, 0.f, 0.f, 0.f, 0.f};
    v8f acc1 = {0.f, 0.f, 0.f, 0.f, 0.f, 0.f, 0.f, 0.f};
#pragma unroll 8
    for (int kc = 0; kc < KCHUNKS; kc += 2) {
      // ---- even chunk -> chain 0 ----
      Frag ah0, al0, bh0, bl0;
      ah0.q[0] = *(const uint4*)(Ah + kc * 1024);
      ah0.q[1] = *(const uint4*)(Ah + kc * 1024 + 16);
      al0.q[0] = *(const uint4*)(Al + kc * 1024);
      al0.q[1] = *(const uint4*)(Al + kc * 1024 + 16);
      bh0.q[0] = *(const uint4*)(wb + kc * 2048);
      bh0.q[1] = *(const uint4*)(wb + kc * 2048 + 16);
      bl0.q[0] = *(const uint4*)(wb + kc * 2048 + 1024);
      bl0.q[1] = *(const uint4*)(wb + kc * 2048 + 1024 + 16);
      // ---- odd chunk -> chain 1 ----
      Frag ah1, al1, bh1, bl1;
      ah1.q[0] = *(const uint4*)(Ah + (kc + 1) * 1024);
      ah1.q[1] = *(const uint4*)(Ah + (kc + 1) * 1024 + 16);
      al1.q[0] = *(const uint4*)(Al + (kc + 1) * 1024);
      al1.q[1] = *(const uint4*)(Al + (kc + 1) * 1024 + 16);
      bh1.q[0] = *(const uint4*)(wb + (kc + 1) * 2048);
      bh1.q[1] = *(const uint4*)(wb + (kc + 1) * 2048 + 16);
      bl1.q[0] = *(const uint4*)(wb + (kc + 1) * 2048 + 1024);
      bl1.q[1] = *(const uint4*)(wb + (kc + 1) * 2048 + 1024 + 16);
      // bf16x3 split-precision fp32 emulation: hi*hi + hi*lo + lo*hi
      acc0 = __builtin_amdgcn_wmma_f32_16x16x32_bf16(false, ah0.v, false, bh0.v,
                                                     (short)0, acc0, false, false);
      acc1 = __builtin_amdgcn_wmma_f32_16x16x32_bf16(false, ah1.v, false, bh1.v,
                                                     (short)0, acc1, false, false);
      acc0 = __builtin_amdgcn_wmma_f32_16x16x32_bf16(false, ah0.v, false, bl0.v,
                                                     (short)0, acc0, false, false);
      acc1 = __builtin_amdgcn_wmma_f32_16x16x32_bf16(false, ah1.v, false, bl1.v,
                                                     (short)0, acc1, false, false);
      acc0 = __builtin_amdgcn_wmma_f32_16x16x32_bf16(false, al0.v, false, bh0.v,
                                                     (short)0, acc0, false, false);
      acc1 = __builtin_amdgcn_wmma_f32_16x16x32_bf16(false, al1.v, false, bh1.v,
                                                     (short)0, acc1, false, false);
    }
    const float bv    = bias[nbase + col];
    const int   mycol = nbase + col;
#pragma unroll
    for (int i = 0; i < 8; ++i) {
      float v = acc0[i] + acc1[i] + bv;
      if (lab[i] == mycol) t_lds[msub * 16 + mrow + i] = v;  // unique writer
      float mn = fmaxf(m_run[i], v);
      s_run[i] = s_run[i] * __expf(m_run[i] - mn) + __expf(v - mn);
      m_run[i] = mn;
    }
  }

#pragma unroll
  for (int i = 0; i < 8; ++i) {
    stat_m[wave * 256 + lane * 8 + i] = m_run[i];
    stat_s[wave * 256 + lane * 8 + i] = s_run[i];
  }
  __syncthreads();

  // ---- combine 64 partials per token (4 waves x 16 lanes), fixed order ----
  {
    const int t  = tid >> 3;        // token-in-tile 0..31
    const int p0 = (tid & 7) * 8;   // this thread's 8 partials
    const int msubT = t >> 4, rt = t & 15;
    const int laneBase = (rt >= 8) ? 16 : 0;
    const int ii = rt & 7;
    float mm = -3.0e38f, ss = 0.0f;
    for (int p = p0; p < p0 + 8; ++p) {
      int w = msubT * 4 + (p >> 4);
      int l = laneBase + (p & 15);
      merge_ms(mm, ss, stat_m[w * 256 + l * 8 + ii], stat_s[w * 256 + l * 8 + ii]);
    }
    red_m[tid] = mm;
    red_s[tid] = ss;
  }
  __syncthreads();
  if ((tid & 7) == 0) {
    const int t = tid >> 3;
    float M = -3.0e38f, S = 0.0f;
    for (int j = 0; j < 8; ++j) merge_ms(M, S, red_m[tid + j], red_s[tid + j]);
    const int o = slice * TOKENS + (m0 + t);
    m_part[o] = M;
    s_part[o] = S;
    t_part[o] = t_lds[t];
  }
}

// ---------------------------------------------------------------------------
// Kernel 3: deterministic merge of slice partials + weighted-mean reduction.
// ---------------------------------------------------------------------------
__global__ __launch_bounds__(256) void finalize_kernel(
    const int* __restrict__ labels, const float* __restrict__ lw,
    const float* __restrict__ m_part, const float* __restrict__ s_part,
    const float* __restrict__ t_part, float* __restrict__ out) {
  __shared__ float sm1[256], sm2[256];
  const int tid = threadIdx.x;
  float wn = 0.0f, wl = 0.0f;
  for (int tok = tid; tok < TOKENS; tok += 256) {
    float M = m_part[tok], S = s_part[tok];
    for (int sl = 1; sl < NSLICE; ++sl)
      merge_ms(M, S, m_part[sl * TOKENS + tok], s_part[sl * TOKENS + tok]);
    int   lb   = labels[tok];
    float tgt  = t_part[(lb / VPS) * TOKENS + tok];
    float logZ = M + __logf(S);
    float w    = lw[tok];
    wn += w * (logZ - tgt);
    wl += w;
  }
  sm1[tid] = wn;
  sm2[tid] = wl;
  __syncthreads();
  for (int st = 128; st > 0; st >>= 1) {
    if (tid < st) { sm1[tid] += sm1[tid + st]; sm2[tid] += sm2[tid + st]; }
    __syncthreads();
  }
  if (tid == 0) out[0] = sm1[0] / sm2[0];
}

extern "C" void kernel_launch(void* const* d_in, const int* in_sizes, int n_in,
                              void* d_out, int out_size, void* d_ws,
                              size_t ws_size, hipStream_t stream) {
  const float* H      = (const float*)d_in[0];  // [4096,1024]
  const float* W      = (const float*)d_in[1];  // [32000,1024]
  const float* bias   = (const float*)d_in[2];  // [32000]
  const int*   labels = (const int*)d_in[3];    // [4096]
  const float* lw     = (const float*)d_in[4];  // [4096]
  float*       out    = (float*)d_out;

  unsigned char* ws     = (unsigned char*)d_ws;
  unsigned char* Wp     = ws;
  float*         m_part = (float*)(ws + WP_BYTES);
  float*         s_part = m_part + NSLICE * TOKENS;
  float*         t_part = s_part + NSLICE * TOKENS;

  wprep_kernel<<<16000, 256, 0, stream>>>(W, Wp);
  dim3 grid(NTILES_TOK, NSLICE);
  celoss_main_kernel<<<grid, 256, 0, stream>>>(H, bias, labels, Wp, m_part,
                                               s_part, t_part);
  finalize_kernel<<<1, 256, 0, stream>>>(labels, lw, m_part, s_part, t_part,
                                         out);
}
